// DotProductAttention_47811575939373
// MI455X (gfx1250) — matrix operand order repaired and checked by reading, not compiled, not run on hardware
//
#include <hip/hip_runtime.h>

typedef __attribute__((ext_vector_type(16))) _Float16 v16h;
typedef __attribute__((ext_vector_type(8)))  _Float16 v8h;
typedef __attribute__((ext_vector_type(4)))  _Float16 v4h;
typedef __attribute__((ext_vector_type(8)))  float    v8f;
typedef __attribute__((ext_vector_type(4)))  float    v4f;

namespace {
constexpr int kB   = 8;
constexpr int kLQ  = 2048;
constexpr int kLK  = 2048;
constexpr int kD   = 64;
constexpr int kTK  = 64;              // keys per LDS tile
constexpr int kQW  = 16;              // queries per wave
constexpr int kNW  = 4;               // waves per block
constexpr int kQB  = kQW * kNW;       // 64 queries per block
constexpr int kPad = 8;               // halves of padding per LDS row
constexpr int kLdK = kD  + kPad;      // 72 halves -> 144B row stride
constexpr int kLdV = kTK + kPad;      // 72 halves -> 144B row stride
constexpr int kBufH = kTK * kLdK + kD * kLdV;   // halves per LDS tile buffer
}

union HF16 { v16h v; v8h h[2]; };

static __device__ __forceinline__ v8f wmma_f16(v16h a, v16h b, v8f c) {
  // D = A(16x32 f16) * B(32x16 f16) + C(16x16 f32)
  return __builtin_amdgcn_wmma_f32_16x16x32_f16(false, a, false, b,
                                                (short)0, c, false, false);
}

static __device__ __forceinline__ unsigned lds_off(const void* p) {
  // generic LDS pointer: low 32 bits = allocation-relative LDS byte offset
  return (unsigned)(uintptr_t)p;
}

static __device__ __forceinline__ void async_b128(unsigned lds, unsigned long long gaddr) {
  asm volatile("global_load_async_to_lds_b128 %0, %1, off"
               :: "v"(lds), "v"(gaddr) : "memory");
}
static __device__ __forceinline__ void wait_async0() {
  asm volatile("s_wait_asynccnt 0x0" ::: "memory");
}

// ---------------- shared compute core (no staging, no barriers) -------------
template<bool MASKED>
static __device__ __forceinline__ void compute_tile(
    int k0, int vlen,
    const _Float16* kbuf, const _Float16* vbuf,
    const HF16 (&qf)[2], v8f (&oacc)[4],
    float& m_run, float& l_run, int nn, int hi, int p8)
{
  // ---- GEMM1: four 16-key S^T tiles (A = K-tile, B = Q^T) ----
  v8f cst[4];
  #pragma unroll
  for (int jt = 0; jt < 4; ++jt) {
    const _Float16* kr = &kbuf[(jt * 16 + nn) * kLdK + hi * 16];
    HF16 af0, af1;
    af0.h[0] = *(const v8h*)(kr);
    af0.h[1] = *(const v8h*)(kr + 8);
    af1.h[0] = *(const v8h*)(kr + 32);
    af1.h[1] = *(const v8h*)(kr + 40);
    v8f acc = {};
    acc = wmma_f16(af0.v, qf[0].v, acc);
    acc = wmma_f16(af1.v, qf[1].v, acc);
    cst[jt] = acc;
  }

  // ---- one online-softmax round over the 64-key tile ----
  float sc[4][8];
  float cmax = -3.0e38f;
  #pragma unroll
  for (int jt = 0; jt < 4; ++jt)
    #pragma unroll
    for (int v = 0; v < 8; ++v) {
      float x = cst[jt][v];
      if (MASKED) {
        int key = k0 + jt * 16 + p8 + v;
        x = (key < vlen) ? x : -1.0e6f;
      }
      sc[jt][v] = x;
      cmax = fmaxf(cmax, x);
    }
  cmax = fmaxf(cmax, __shfl_xor(cmax, 16, 32));
  float mold  = m_run;
  float mnew  = fmaxf(mold, cmax);
  float alpha = __expf(mold - mnew);
  m_run = mnew;

  if (__any(mnew > mold)) {          // rescale O only when max grew
    #pragma unroll
    for (int v = 0; v < 8; ++v) {
      float av = __shfl(alpha, v + p8, 32);
      #pragma unroll
      for (int t = 0; t < 4; ++t) oacc[t][v] *= av;
    }
  }

  float rsum = 0.0f;
  HF16 pf[2];                        // A-layout P fragments (2 x 32 keys)
  #pragma unroll
  for (int jt = 0; jt < 4; ++jt)
    #pragma unroll
    for (int v = 0; v < 8; ++v) {
      float e = __expf(sc[jt][v] - mnew);
      rsum += e;
      pf[jt >> 1].h[jt & 1][v] = (_Float16)e;
    }
  rsum += __shfl_xor(rsum, 16, 32);
  l_run = l_run * alpha + rsum;

  // ---- GEMM2: O += P x V (2 key-chunks x 4 d-tiles) ----
  #pragma unroll
  for (int cc = 0; cc < 2; ++cc)
    #pragma unroll
    for (int t = 0; t < 4; ++t) {
      const _Float16* vr = &vbuf[(t * 16 + nn) * kLdV + cc * 32 + p8];
      HF16 bf;
      bf.h[0] = *(const v8h*)(vr);
      bf.h[1] = *(const v8h*)(vr + 16);
      oacc[t] = wmma_f16(pf[cc].v, bf.v, oacc[t]);
    }
}

// ---------------- pre-pass: K,V fp32 -> f16 (V transposed per batch) --------
__global__ __launch_bounds__(256)
void convert_kv(const float* __restrict__ K, const float* __restrict__ V,
                _Float16* __restrict__ Kh, _Float16* __restrict__ Vt)
{
  __shared__ _Float16 tl[kTK * kLdK];          // V tile staging for transpose
  const int b  = blockIdx.y;
  const int k0 = blockIdx.x * kTK;
  const int tid = threadIdx.x;
  const float*    kb  = K  + ((size_t)b * kLK + k0) * kD;
  const float*    vb  = V  + ((size_t)b * kLK + k0) * kD;
  _Float16*       khb = Kh + ((size_t)b * kLK + k0) * kD;
  _Float16*       vtb = Vt + (size_t)b * kD * kLK + k0;

  #pragma unroll
  for (int r = 0; r < 4; ++r) {                // 1024 float4 groups / 256 thr
    int idx = r * 256 + tid;
    int kk  = idx >> 4;
    int dg  = (idx & 15) << 2;
    v4f kv = *(const v4f*)(kb + kk * kD + dg);
    v4f vv = *(const v4f*)(vb + kk * kD + dg);
    v4h kh, vh;
    #pragma unroll
    for (int j = 0; j < 4; ++j) { kh[j] = (_Float16)kv[j]; vh[j] = (_Float16)vv[j]; }
    *(v4h*)(khb + kk * kD + dg) = kh;          // K: straight f16 copy
    *(v4h*)(&tl[kk * kLdK + dg]) = vh;         // V: stage for transpose
  }
  __syncthreads();
  #pragma unroll
  for (int r = 0; r < 4; ++r) {                // write Vt[d][k] coalesced in k
    int idx = r * 256 + tid;
    int d   = idx >> 4;
    int kk4 = (idx & 15) << 2;
    v4h o;
    #pragma unroll
    for (int j = 0; j < 4; ++j) o[j] = tl[(kk4 + j) * kLdK + d];
    *(v4h*)(vtb + (size_t)d * kLK + kk4) = o;
  }
}

// ---------------- main kernel: async-staged, double-buffered ----------------
static __device__ __forceinline__ void stage_issue(
    const _Float16* khb,   // global K tile base (row stride kD*2 bytes)
    const _Float16* vtb,   // global Vt base + k0 (row stride kLK*2 bytes)
    _Float16* kbuf, _Float16* vbuf, int tid)
{
  const int row = tid >> 3;                    // 0..15, +16 per step
  const int col = tid & 7;                     // 16B column
  unsigned kl = lds_off(kbuf) + row * (kLdK * 2) + col * 16;
  unsigned vl = lds_off(vbuf) + row * (kLdV * 2) + col * 16;
  unsigned long long kg = (unsigned long long)(uintptr_t)khb + row * (kD * 2) + col * 16;
  unsigned long long vg = (unsigned long long)(uintptr_t)vtb + (size_t)row * (kLK * 2) + col * 16;
  #pragma unroll
  for (int r = 0; r < 4; ++r) {
    async_b128(kl, kg);
    async_b128(vl, vg);
    kl += 16 * (kLdK * 2); kg += 16 * (kD * 2);
    vl += 16 * (kLdV * 2); vg += (unsigned long long)16 * (kLK * 2);
  }
}

__global__ __launch_bounds__(kNW * 32)
void flash_attn_fwd_f16wmma_pre(const float* __restrict__ Qg,
                                const _Float16* __restrict__ Kh,
                                const _Float16* __restrict__ Vt,
                                const int* __restrict__ vlens,
                                float* __restrict__ Og)
{
  __shared__ __align__(16) _Float16 smem[2 * kBufH];
  _Float16* kbuf0 = smem;
  _Float16* vbuf0 = smem + kTK * kLdK;
  _Float16* kbuf1 = smem + kBufH;
  _Float16* vbuf1 = smem + kBufH + kTK * kLdK;

  const int tid  = threadIdx.x;
  const int lane = tid & 31;
  const int wave = tid >> 5;
  const int nn   = lane & 15;
  const int hi   = lane >> 4;
  const int p8   = hi * 8;

  const int b    = blockIdx.y;
  const int vlen = vlens[b];
  const size_t qrow0 = (size_t)b * kLQ + (size_t)blockIdx.x * kQB + wave * kQW;

  const _Float16* Khb = Kh + (size_t)b * kLK * kD;
  const _Float16* Vtb = Vt + (size_t)b * kD * kLK;

  const int  ntiles  = (vlen + kTK - 1) / kTK;
  const bool partial = (vlen & (kTK - 1)) != 0;

  stage_issue(Khb, Vtb, kbuf0, vbuf0, tid);     // prefetch tile 0

  // ---- Q fragments (B-layout, pre-scaled by 1/sqrt(D)) ----
  HF16 qf[2];
  {
    const float* qp = Qg + (qrow0 + nn) * kD + hi * 16;
    #pragma unroll
    for (int c = 0; c < 2; ++c)
      #pragma unroll
      for (int i = 0; i < 16; ++i)
        qf[c].v[i] = (_Float16)(qp[c * 32 + i] * 0.125f);
  }

  v8f oacc[4] = {};
  float m_run = -3.0e38f;
  float l_run = 0.0f;

  for (int it = 0; it < ntiles; ++it) {
    wait_async0();                 // my portion of current tile is in LDS
    __syncthreads();               // everyone's portion is in LDS
    const _Float16* kb = (it & 1) ? kbuf1 : kbuf0;
    const _Float16* vb = (it & 1) ? vbuf1 : vbuf0;
    if (it + 1 < ntiles) {         // prefetch next tile into the other buffer
      int nk0 = (it + 1) * kTK;
      stage_issue(Khb + (size_t)nk0 * kD, Vtb + nk0,
                  (it & 1) ? kbuf0 : kbuf1, (it & 1) ? vbuf0 : vbuf1, tid);
    }
    if (partial && it == ntiles - 1)
      compute_tile<true >(it * kTK, vlen, kb, vb, qf, oacc, m_run, l_run, nn, hi, p8);
    else
      compute_tile<false>(it * kTK, vlen, kb, vb, qf, oacc, m_run, l_run, nn, hi, p8);
  }

  float rl = 1.0f / l_run;
  #pragma unroll
  for (int v = 0; v < 8; ++v) {
    float rv = __shfl(rl, v + p8, 32);
    float* orow = Og + (qrow0 + p8 + v) * kD + nn;
    #pragma unroll
    for (int t = 0; t < 4; ++t)
      orow[t * 16] = oacc[t][v] * rv;
  }
}

// ---------------- fallback (self-contained, converts in-kernel) -------------
template<bool MASKED>
static __device__ __forceinline__ void process_tile_conv(
    int k0, int vlen,
    const float* __restrict__ kbase, const float* __restrict__ vbase,
    _Float16* k_lds, _Float16* vt_lds,
    const HF16 (&qf)[2], v8f (&oacc)[4],
    float& m_run, float& l_run, int tid, int nn, int hi, int p8)
{
  __syncthreads();
  #pragma unroll
  for (int r = 0; r < (kTK * kD / 4) / (kNW * 32); ++r) {
    int idx = r * (kNW * 32) + tid;
    int kk  = idx >> 4;
    int dg  = (idx & 15) << 2;
    v4f kv = *(const v4f*)(kbase + (size_t)(k0 + kk) * kD + dg);
    v4f vv = *(const v4f*)(vbase + (size_t)(k0 + kk) * kD + dg);
    #pragma unroll
    for (int j = 0; j < 4; ++j) {
      k_lds[kk * kLdK + dg + j]    = (_Float16)kv[j];
      vt_lds[(dg + j) * kLdV + kk] = (_Float16)vv[j];
    }
  }
  __syncthreads();
  compute_tile<MASKED>(k0, vlen, k_lds, vt_lds, qf, oacc, m_run, l_run, nn, hi, p8);
}

__global__ __launch_bounds__(kNW * 32)
void flash_attn_fwd_f16wmma_conv(const float* __restrict__ Qg,
                                 const float* __restrict__ Kg,
                                 const float* __restrict__ Vg,
                                 const int*   __restrict__ vlens,
                                 float* __restrict__ Og)
{
  __shared__ _Float16 k_lds[kTK * kLdK];
  __shared__ _Float16 vt_lds[kD * kLdV];

  const int tid  = threadIdx.x;
  const int lane = tid & 31;
  const int wave = tid >> 5;
  const int nn   = lane & 15;
  const int hi   = lane >> 4;
  const int p8   = hi * 8;

  const int b    = blockIdx.y;
  const int vlen = vlens[b];
  const size_t qrow0 = (size_t)b * kLQ + (size_t)blockIdx.x * kQB + wave * kQW;

  HF16 qf[2];
  {
    const float* qp = Qg + (qrow0 + nn) * kD + hi * 16;
    #pragma unroll
    for (int c = 0; c < 2; ++c)
      #pragma unroll
      for (int i = 0; i < 16; ++i)
        qf[c].v[i] = (_Float16)(qp[c * 32 + i] * 0.125f);
  }

  v8f oacc[4] = {};
  float m_run = -3.0e38f;
  float l_run = 0.0f;

  const float* kbase = Kg + (size_t)b * kLK * kD;
  const float* vbase = Vg + (size_t)b * kLK * kD;

  const int nfull = vlen / kTK;
  for (int it = 0; it < nfull; ++it)
    process_tile_conv<false>(it * kTK, vlen, kbase, vbase, k_lds, vt_lds,
                             qf, oacc, m_run, l_run, tid, nn, hi, p8);
  if (vlen & (kTK - 1))
    process_tile_conv<true>(nfull * kTK, vlen, kbase, vbase, k_lds, vt_lds,
                            qf, oacc, m_run, l_run, tid, nn, hi, p8);

  float rl = 1.0f / l_run;
  #pragma unroll
  for (int v = 0; v < 8; ++v) {
    float rv = __shfl(rl, v + p8, 32);
    float* orow = Og + (qrow0 + p8 + v) * kD + nn;
    #pragma unroll
    for (int t = 0; t < 4; ++t)
      orow[t * 16] = oacc[t][v] * rv;
  }
}

extern "C" void kernel_launch(void* const* d_in, const int* in_sizes, int n_in,
                              void* d_out, int out_size, void* d_ws, size_t ws_size,
                              hipStream_t stream) {
  (void)in_sizes; (void)n_in; (void)out_size;
  const float* Q  = (const float*)d_in[0];
  const float* K  = (const float*)d_in[1];
  const float* V  = (const float*)d_in[2];
  const int*   vl = (const int*)d_in[3];
  float* out = (float*)d_out;

  dim3 grid(kLQ / kQB, kB);
  dim3 block(kNW * 32);

  const size_t halvesKV = (size_t)kB * kLK * kD;      // per matrix
  if (ws_size >= 2 * halvesKV * sizeof(_Float16)) {
    _Float16* Kh = (_Float16*)d_ws;
    _Float16* Vt = Kh + halvesKV;
    convert_kv<<<dim3(kLK / kTK, kB), 256, 0, stream>>>(K, V, Kh, Vt);
    flash_attn_fwd_f16wmma_pre<<<grid, block, 0, stream>>>(Q, Kh, Vt, vl, out);
  } else {
    flash_attn_fwd_f16wmma_conv<<<grid, block, 0, stream>>>(Q, K, V, vl, out);
  }
}
